// IndexConvMSWA_48017734369556
// MI455X (gfx1250) — compile-verified, matrix operand order checked by reading, not run on hardware
//
#include <hip/hip_runtime.h>
#include <hip/hip_bf16.h>
#include <math.h>

typedef __attribute__((ext_vector_type(16))) __bf16 v16bf;
typedef __attribute__((ext_vector_type(8)))  float  v8f;

#define NVOX (64*64*64)          // 262144 voxels
#define NCH  128
#define GN_N (16.0f * (float)NVOX) // elements per group (16 ch * NVOX)

// ---------------- workspace layout (float offsets) ----------------
#define OFF_CONV   0                       // 128*262144 f32  (conv1 output, pre-GN)
#define FB         (NCH*NVOX)              // 33554432
#define OFF_POOL   (FB + 0)                // [c][cell] 128*512
#define OFF_QS     (FB + 65536)            // [cell][c] 512*128  pooled+GN1
#define OFF_QG     (FB + 131072)           // [cell][c] qs + posenc
#define OFF_PF     (FB + 196608)           // [row][c] 256*128
#define OFF_KM     (FB + 229376)           // 256*128
#define OFF_VM     (FB + 262144)           // 256*128
#define OFF_QM     (FB + 294912)           // 512*128
#define OFF_QS2    (FB + 360448)           // qs + attn out
#define OFF_H1     (FB + 425984)           // 512*256
#define OFF_QSF    (FB + 557056)           // 512*128
#define OFF_Y      (FB + 622592)           // [c][cell] 128*512  (out_w@qsf + out_b)
#define OFF_AB1    (FB + 688128)           // a1[128], b1[128]
#define OFF_AB2    (FB + 688384)           // a2[128], b2[128]
#define OFF_P1     (FB + 688640)           // 1024 sums + 1024 sumsq  (GN1 partials)
#define OFF_P2     (FB + 690688)           // 4096*8 sums + 4096*8 sumsq (GN2 partials)

// =================================================================
// K1: conv1x1 64->128 over all voxels, bf16 WMMA, f32 accumulate.
//   C[128 x 262144] = W[128x64] @ X[64x262144] + bias
// block = 256 threads (8 waves). Each block does one 64-voxel tile.
// All 8 B-fragment ds_load pairs are issued before the WMMA chain so
// the dscnt waits can be partial and LDS latency overlaps the XDL pipe.
// =================================================================
__global__ __launch_bounds__(256) void k_conv_in(const float* __restrict__ x,
                                                 const float* __restrict__ w,
                                                 const float* __restrict__ bias,
                                                 float* __restrict__ out) {
  __shared__ __bf16 lx[64][72];            // [voxel][cin], padded rows (144B, 16B-aligned)
  const int tid  = threadIdx.x;
  const int wave = tid >> 5;               // 0..7 (wave32)
  const int lane = tid & 31;
  const int m0   = wave * 16;              // output-channel tile base
  const int n0   = blockIdx.x * 64;        // voxel base

  // ---- A fragments (per ISA 7.12.2, 16-bit A 16x32): built once, reused ----
  const int am = lane & 15, ahi = lane >> 4;
  v16bf a0, a1;
  {
    const float* wr = w + (m0 + am) * 64;
#pragma unroll
    for (int t = 0; t < 16; ++t) {
      int k = ((t & 8) ? 16 : 0) + ahi * 8 + (t & 7);
      a0[t] = (__bf16)wr[k];
      a1[t] = (__bf16)wr[32 + k];
    }
  }

  // ---- stage 64cin x 64vox tile to LDS (transposed, bf16) ----
  {
    int v  = tid & 63;
    int c4 = tid >> 6;                      // 0..3
#pragma unroll
    for (int i = 0; i < 16; ++i) {
      int cin = c4 * 16 + i;
      lx[v][cin] = (__bf16)x[cin * NVOX + n0 + v];
      // warm L2 for the tile 4 blocks ahead (same cin rows, +256 voxels)
      __builtin_prefetch(x + cin * NVOX + n0 + v + 256, 0, 1);
    }
  }
  __syncthreads();

  const int bn = lane & 15, bhi = lane >> 4;

  // issue ALL B-fragment LDS reads first (4 subtiles x 2 K-halves)
  v16bf bf[4][2];
#pragma unroll
  for (int nt = 0; nt < 4; ++nt) {
    const __bf16* row = &lx[nt * 16 + bn][0];
#pragma unroll
    for (int t = 0; t < 16; ++t) {
      int k = ((t & 8) ? 16 : 0) + bhi * 8 + (t & 7);
      bf[nt][0][t] = row[k];
      bf[nt][1][t] = row[32 + k];
    }
  }

  // WMMA chain
  v8f acc[4];
#pragma unroll
  for (int nt = 0; nt < 4; ++nt) {
    v8f c = {};
    c = __builtin_amdgcn_wmma_f32_16x16x32_bf16(false, a0, false, bf[nt][0], (short)0, c, false, false);
    c = __builtin_amdgcn_wmma_f32_16x16x32_bf16(false, a1, false, bf[nt][1], (short)0, c, false, false);
    acc[nt] = c;
  }

  // C/D layout: VGPR r -> M = r (lanes 0-15) / r+8 (lanes 16-31), N = lane&15
#pragma unroll
  for (int nt = 0; nt < 4; ++nt) {
#pragma unroll
    for (int r = 0; r < 8; ++r) {
      int ch = m0 + r + bhi * 8;
      out[ch * NVOX + n0 + nt * 16 + bn] = acc[nt][r] + bias[ch];
    }
  }
}

// =================================================================
// K2: per-(channel, x-block) reduction: 8^3 pool sums + GN1 partials
// grid = 128*8 blocks, 256 threads
// =================================================================
__global__ __launch_bounds__(256) void k_reduce1(const float* __restrict__ conv,
                                                 float* __restrict__ pool,
                                                 float* __restrict__ p1) {
  const int b = blockIdx.x;
  const int c = b >> 3, bx = b & 7;
  const int t = threadIdx.x;
  const int cell = t & 63, sub = t >> 6;    // cell=(by,bz), 4 sub-threads per cell
  const int by = cell >> 3, bz = cell & 7;
  const float* p = conv + c * NVOX + (bx * 8) * 4096 + (by * 8) * 64 + bz * 8;
  float s = 0.f, s2 = 0.f;
  for (int xp = 0; xp < 8; ++xp)
    for (int yp = sub * 2; yp < sub * 2 + 2; ++yp)
#pragma unroll
      for (int zp = 0; zp < 8; ++zp) {
        float v = p[xp * 4096 + yp * 64 + zp];
        s += v; s2 += v * v;
      }
  __shared__ float lp[256], lq[256];
  lp[t] = s; lq[t] = s2;
  __syncthreads();
  if (sub == 0)
    pool[c * 512 + bx * 64 + cell] = lp[cell] + lp[cell + 64] + lp[cell + 128] + lp[cell + 192];
  for (int o = 128; o > 0; o >>= 1) {
    if (t < o) { lp[t] += lp[t + o]; lq[t] += lq[t + o]; }
    __syncthreads();
  }
  if (t == 0) { p1[b] = lp[0]; p1[1024 + b] = lq[0]; }
}

// =================================================================
// K3: finish GN1 (a1,b1), build q_s (pooled, GN'd) and qg (+pos enc)
// single block, 256 threads
// =================================================================
__global__ __launch_bounds__(256) void k_gn1_pool(const float* __restrict__ p1,
                                                  const float* __restrict__ gn1w,
                                                  const float* __restrict__ gn1b,
                                                  const float* __restrict__ pool,
                                                  float* __restrict__ ab1,
                                                  float* __restrict__ qs,
                                                  float* __restrict__ qg) {
  __shared__ float sgm[8], sgv[8], sa[128], sb[128];
  const int t = threadIdx.x;
  if (t < 8) {
    float s = 0.f, q = 0.f;
    for (int i = 0; i < 128; ++i) { s += p1[t * 128 + i]; q += p1[1024 + t * 128 + i]; }
    float m = s / GN_N;
    sgm[t] = m; sgv[t] = q / GN_N - m * m;
  }
  __syncthreads();
  if (t < 128) {
    int g = t >> 4;
    float a = gn1w[t] * rsqrtf(sgv[g] + 1e-5f);
    float b = gn1b[t] - sgm[g] * a;
    sa[t] = a; sb[t] = b; ab1[t] = a; ab1[128 + t] = b;
  }
  __syncthreads();
  for (int i = t; i < 512 * 128; i += 256) {
    int cell = i >> 7, c = i & 127;
    float v = pool[c * 512 + cell] * (1.0f / 512.0f) * sa[c] + sb[c];
    qs[i] = v;
    float pe = 0.f;
    if (c < 126) {                  // d = 128//3 = 42 per axis, 2 zero pad
      int axis = c / 42, j = c - axis * 42;
      int p = (axis == 0) ? (cell >> 6) : ((axis == 1) ? ((cell >> 3) & 7) : (cell & 7));
      float pos  = (float)(p + 1) * (6.2831853071f / 8.0f);
      float expn = (float)(2 * (j >> 1)) / 42.0f;
      float arg  = pos * __powf(10000.0f, -expn);
      pe = (j & 1) ? __cosf(arg) : __sinf(arg);
    }
    qg[i] = v + pe;
  }
}

// ---------------- small dense GEMM helpers (fp32, tiny sizes) ----------------
__global__ __launch_bounds__(256) void k_pf(const float* __restrict__ part_f,
                                            const float* __restrict__ pw,
                                            const float* __restrict__ pb,
                                            float* __restrict__ pf) {
  int i = blockIdx.x * 256 + threadIdx.x;   // 256*128
  int row = i >> 7, c = i & 127;
  const float* a = part_f + row * 128;
  const float* w = pw + c * 128;
  float s = pb[c];
  for (int k = 0; k < 128; ++k) s += a[k] * w[k];
  pf[i] = s;
}

__global__ __launch_bounds__(256) void k_kv(const float* __restrict__ pf,
                                            const float* __restrict__ kw, const float* __restrict__ kb,
                                            const float* __restrict__ vw, const float* __restrict__ vb,
                                            float* __restrict__ km, float* __restrict__ vm) {
  int b = blockIdx.x;
  bool isv = b >= 128;
  int i = (isv ? b - 128 : b) * 256 + threadIdx.x;
  int row = i >> 7, c = i & 127;
  const float* w = (isv ? vw : kw) + c * 128;
  const float* a = pf + row * 128;
  float s = (isv ? vb : kb)[c];
  for (int k = 0; k < 128; ++k) s += a[k] * w[k];
  (isv ? vm : km)[i] = s;
}

__global__ __launch_bounds__(256) void k_q(const float* __restrict__ qg,
                                           const float* __restrict__ qw, const float* __restrict__ qb,
                                           float* __restrict__ qm) {
  int i = blockIdx.x * 256 + threadIdx.x;   // 512*128
  int row = i >> 7, c = i & 127;
  const float* a = qg + row * 128;
  const float* w = qw + c * 128;
  float s = qb[c];
  for (int k = 0; k < 128; ++k) s += a[k] * w[k];
  qm[i] = s;
}

// =================================================================
// K4: attention, 4 heads x 512 queries x 256 parts, hd=32 (no sqrt(d) scaling)
// grid = 4*32 blocks (16 queries each), 256 threads
// =================================================================
__global__ __launch_bounds__(256) void k_attn(const float* __restrict__ qm,
                                              const float* __restrict__ km,
                                              const float* __restrict__ vm,
                                              const float* __restrict__ qs,
                                              float* __restrict__ qs2) {
  __shared__ float sc[16][260];
  __shared__ float red[16][17];
  __shared__ float rowstat[16];
  const int h = blockIdx.x >> 5;
  const int qb = blockIdx.x & 31;
  const int t = threadIdx.x;
  const int ql = t & 15, ck = t >> 4;
  const int qrow = qb * 16 + ql;
  const float* qp = qm + qrow * 128 + h * 32;
  float mx = -3.4e38f;
  for (int pp = 0; pp < 16; ++pp) {
    const float* kp = km + (ck * 16 + pp) * 128 + h * 32;
    float s = 0.f;
#pragma unroll
    for (int d = 0; d < 32; ++d) s += qp[d] * kp[d];
    sc[ql][ck * 16 + pp] = s;
    mx = fmaxf(mx, s);
  }
  red[ql][ck] = mx;
  __syncthreads();
  if (ck == 0) {
    float m = red[ql][0];
    for (int i = 1; i < 16; ++i) m = fmaxf(m, red[ql][i]);
    rowstat[ql] = m;
  }
  __syncthreads();
  mx = rowstat[ql];
  float sum = 0.f;
  for (int pp = 0; pp < 16; ++pp) {
    int p = ck * 16 + pp;
    float e = __expf(sc[ql][p] - mx);
    sc[ql][p] = e; sum += e;
  }
  red[ql][ck] = sum;
  __syncthreads();
  if (ck == 0) {
    float s = 0.f;
    for (int i = 0; i < 16; ++i) s += red[ql][i];
    rowstat[ql] = 1.0f / s;
  }
  __syncthreads();
  for (int r = 0; r < 2; ++r) {
    int idx = t + 256 * r;                  // 16q x 32d
    int oq = idx >> 5, d = idx & 31;
    const float* vp = vm + h * 32 + d;
    float acc = 0.f;
    for (int p = 0; p < 256; ++p) acc += sc[oq][p] * vp[p * 128];
    acc *= rowstat[oq];
    int cell = qb * 16 + oq, c = h * 32 + d;
    qs2[cell * 128 + c] = qs[cell * 128 + c] + acc;
  }
}

// ---------------- MLP (gelu tanh-approx, as jax.nn.gelu default) ----------------
__device__ __forceinline__ float gelu_t(float x) {
  return 0.5f * x * (1.0f + tanhf(0.7978845608f * (x + 0.044715f * x * x * x)));
}

__global__ __launch_bounds__(256) void k_mlp1(const float* __restrict__ qs2,
                                              const float* __restrict__ w, const float* __restrict__ b,
                                              float* __restrict__ h1) {
  int i = blockIdx.x * 256 + threadIdx.x;   // 512*256
  int cell = i >> 8, j = i & 255;
  const float* a = qs2 + cell * 128;
  const float* wr = w + j * 128;
  float s = b[j];
  for (int k = 0; k < 128; ++k) s += a[k] * wr[k];
  h1[i] = gelu_t(s);
}

__global__ __launch_bounds__(256) void k_mlp2(const float* __restrict__ h1,
                                              const float* __restrict__ w, const float* __restrict__ b,
                                              const float* __restrict__ qs2,
                                              float* __restrict__ qsf) {
  int i = blockIdx.x * 256 + threadIdx.x;   // 512*128
  int cell = i >> 7, c = i & 127;
  const float* a = h1 + cell * 256;
  const float* wr = w + c * 256;
  float s = b[c];
  for (int k = 0; k < 256; ++k) s += a[k] * wr[k];
  qsf[i] = qs2[i] + s;
}

// out_w @ qsf + out_b, stored channel-major: y[c][cell]
__global__ __launch_bounds__(256) void k_proj(const float* __restrict__ qsf,
                                              const float* __restrict__ ow, const float* __restrict__ ob,
                                              float* __restrict__ y) {
  int i = blockIdx.x * 256 + threadIdx.x;   // 128*512
  int c = i >> 9, cell = i & 511;
  const float* a = qsf + cell * 128;
  const float* wr = ow + c * 128;
  float s = ob[c];
  for (int k = 0; k < 128; ++k) s += a[k] * wr[k];
  y[i] = s;
}

// ---------------- shared interp helper: g[c][sz] = bilinear(x,y) of y ----------------
__device__ __forceinline__ void build_sg(const float* __restrict__ y, float sg[128][8],
                                         int X, int Y, int t) {
  int ix0 = X / 9, iy0 = Y / 9;
  float fx = X / 9.0f - (float)ix0, fy = Y / 9.0f - (float)iy0;
  int ix1 = min(ix0 + 1, 7), iy1 = min(iy0 + 1, 7);
  float w00 = (1.f - fx) * (1.f - fy), w10 = fx * (1.f - fy);
  float w01 = (1.f - fx) * fy, w11 = fx * fy;
  for (int i = t; i < 1024; i += 256) {
    int c = i >> 3, sz = i & 7;
    const float* yc = y + c * 512;
    sg[c][sz] = w00 * yc[ix0 * 64 + iy0 * 8 + sz] + w10 * yc[ix1 * 64 + iy0 * 8 + sz]
              + w01 * yc[ix0 * 64 + iy1 * 8 + sz] + w11 * yc[ix1 * 64 + iy1 * 8 + sz];
  }
}

// =================================================================
// K5a: GN2 stats over the (virtually) upsampled conv2 output.
// grid = 64*64 (one per (x,y) column), 256 threads. Deterministic tree reduce.
// =================================================================
__global__ __launch_bounds__(256) void k_stats2(const float* __restrict__ y,
                                                float* __restrict__ p2) {
  __shared__ float sg[128][8];
  __shared__ float rs[256], rq[256];
  const int b = blockIdx.x;
  const int X = b >> 6, Y = b & 63;
  const int t = threadIdx.x;
  build_sg(y, sg, X, Y, t);
  __syncthreads();
  const int c = t >> 1, zb = (t & 1) * 32;
  float s = 0.f, s2 = 0.f;
  for (int zi = 0; zi < 32; ++zi) {
    int z = zb + zi;
    int iz0 = z / 9;
    float fz = z / 9.0f - (float)iz0;
    int iz1 = min(iz0 + 1, 7);
    float v = sg[c][iz0] * (1.f - fz) + sg[c][iz1] * fz;
    s += v; s2 += v * v;
  }
  rs[t] = s; rq[t] = s2;                      // group = c>>4 = t>>5: contiguous 32-thread segments
  __syncthreads();
  for (int o = 16; o > 0; o >>= 1) {
    if ((t & 31) < o) { rs[t] += rs[t + o]; rq[t] += rq[t + o]; }
    __syncthreads();
  }
  if ((t & 31) == 0) {
    int g = t >> 5;
    p2[b * 8 + g] = rs[t];
    p2[32768 + b * 8 + g] = rq[t];
  }
}

// K5b: finish GN2 affine
__global__ __launch_bounds__(256) void k_ab2(const float* __restrict__ p2,
                                             const float* __restrict__ gn2w,
                                             const float* __restrict__ gn2b,
                                             float* __restrict__ ab2) {
  __shared__ float rs[256], rq[256], gm[8], gv[8];
  const int t = threadIdx.x;
  const int g = t >> 5, l = t & 31;
  float s = 0.f, q = 0.f;
  for (int b = l; b < 4096; b += 32) { s += p2[b * 8 + g]; q += p2[32768 + b * 8 + g]; }
  rs[t] = s; rq[t] = q;
  __syncthreads();
  for (int o = 16; o > 0; o >>= 1) {
    if (l < o) { rs[t] += rs[t + o]; rq[t] += rq[t + o]; }
    __syncthreads();
  }
  if (l == 0) { float m = rs[t] / GN_N; gm[g] = m; gv[g] = rq[t] / GN_N - m * m; }
  __syncthreads();
  if (t < 128) {
    int gg = t >> 4;
    float a = gn2w[t] * rsqrtf(gv[gg] + 1e-5f);
    ab2[t] = a; ab2[128 + t] = gn2b[t] - gm[gg] * a;
  }
}

// =================================================================
// K6: fused final: out = GN1(conv_out) + GN2(trilinear(y))
// grid = 64*64, 256 threads; streams conv_out (L2) -> out (HBM)
// =================================================================
__global__ __launch_bounds__(256) void k_final(const float* __restrict__ y,
                                               const float* __restrict__ conv,
                                               const float* __restrict__ ab1,
                                               const float* __restrict__ ab2,
                                               float* __restrict__ out) {
  __shared__ float sg[128][8];
  const int b = blockIdx.x;
  const int X = b >> 6, Y = b & 63;
  const int t = threadIdx.x;
  build_sg(y, sg, X, Y, t);
  __syncthreads();
  const int c = t >> 1, zb = (t & 1) * 32;
  const float a1c = ab1[c], b1c = ab1[128 + c];
  const float a2c = ab2[c], b2c = ab2[128 + c];
  const float* cp = conv + c * NVOX + X * 4096 + Y * 64 + zb;
  float* op = out + c * NVOX + X * 4096 + Y * 64 + zb;
#pragma unroll
  for (int zi = 0; zi < 32; ++zi) {
    int z = zb + zi;
    int iz0 = z / 9;
    float fz = z / 9.0f - (float)iz0;
    int iz1 = min(iz0 + 1, 7);
    float up = sg[c][iz0] * (1.f - fz) + sg[c][iz1] * fz;
    op[zi] = cp[zi] * a1c + b1c + up * a2c + b2c;
  }
}

// =================================================================
extern "C" void kernel_launch(void* const* d_in, const int* in_sizes, int n_in,
                              void* d_out, int out_size, void* d_ws, size_t ws_size,
                              hipStream_t stream) {
  const float* x      = (const float*)d_in[0];
  const float* part_f = (const float*)d_in[1];
  const float* in_w   = (const float*)d_in[2];
  const float* in_b   = (const float*)d_in[3];
  const float* gn1_w  = (const float*)d_in[4];
  const float* gn1_b  = (const float*)d_in[5];
  const float* part_w = (const float*)d_in[6];
  const float* part_b = (const float*)d_in[7];
  const float* q_w    = (const float*)d_in[8];
  const float* q_b    = (const float*)d_in[9];
  const float* k_w    = (const float*)d_in[10];
  const float* k_b    = (const float*)d_in[11];
  const float* v_w    = (const float*)d_in[12];
  const float* v_b    = (const float*)d_in[13];
  const float* mlp1_w = (const float*)d_in[14];
  const float* mlp1_b = (const float*)d_in[15];
  const float* mlp2_w = (const float*)d_in[16];
  const float* mlp2_b = (const float*)d_in[17];
  const float* out_w  = (const float*)d_in[18];
  const float* out_b  = (const float*)d_in[19];
  const float* gn2_w  = (const float*)d_in[20];
  const float* gn2_b  = (const float*)d_in[21];
  float* out = (float*)d_out;
  float* ws  = (float*)d_ws;

  float* conv = ws + OFF_CONV;
  float* pool = ws + OFF_POOL;
  float* qs   = ws + OFF_QS;
  float* qg   = ws + OFF_QG;
  float* pf   = ws + OFF_PF;
  float* km   = ws + OFF_KM;
  float* vm   = ws + OFF_VM;
  float* qm   = ws + OFF_QM;
  float* qs2  = ws + OFF_QS2;
  float* h1   = ws + OFF_H1;
  float* qsf  = ws + OFF_QSF;
  float* y    = ws + OFF_Y;
  float* ab1  = ws + OFF_AB1;
  float* ab2  = ws + OFF_AB2;
  float* p1   = ws + OFF_P1;
  float* p2   = ws + OFF_P2;

  k_conv_in <<<NVOX / 64, 256, 0, stream>>>(x, in_w, in_b, conv);
  k_reduce1 <<<128 * 8,   256, 0, stream>>>(conv, pool, p1);
  k_gn1_pool<<<1,         256, 0, stream>>>(p1, gn1_w, gn1_b, pool, ab1, qs, qg);
  k_pf      <<<128,       256, 0, stream>>>(part_f, part_w, part_b, pf);
  k_kv      <<<256,       256, 0, stream>>>(pf, k_w, k_b, v_w, v_b, km, vm);
  k_q       <<<256,       256, 0, stream>>>(qg, q_w, q_b, qm);
  k_attn    <<<128,       256, 0, stream>>>(qm, km, vm, qs, qs2);
  k_mlp1    <<<512,       256, 0, stream>>>(qs2, mlp1_w, mlp1_b, h1);
  k_mlp2    <<<256,       256, 0, stream>>>(h1, mlp2_w, mlp2_b, qs2, qsf);
  k_proj    <<<256,       256, 0, stream>>>(qsf, out_w, out_b, y);
  k_stats2  <<<64 * 64,   256, 0, stream>>>(y, p2);
  k_ab2     <<<1,         256, 0, stream>>>(p2, gn2_w, gn2_b, ab2);
  k_final   <<<64 * 64,   256, 0, stream>>>(y, conv, ab1, ab2, out);
}
